// GCODERNN_20933670601178
// MI455X (gfx1250) — matrix operand order repaired and checked by reading, not compiled, not run on hardware
//
#include <hip/hip_runtime.h>
#include <math.h>

typedef __attribute__((ext_vector_type(16))) _Float16 v16h;
typedef __attribute__((ext_vector_type(8)))  _Float16 v8h;
typedef __attribute__((ext_vector_type(8)))  float    v8f;

#define DEV __device__ __forceinline__

// ---------------- problem constants ----------------
constexpr int BB = 32, SS = 100, NN = 68;
constexpr int NP = 80;        // node dim padded to 5 tiles of 16
constexpr int KADJ = 96;      // adj K padded to 3 chunks of 32
constexpr int DR = 256, DH = 512, G3 = 768;
constexpr float DTC = 0.05f;

// ---------------- workspace layout (halves) ----------------
constexpr size_t WS_W0T = 0;                      // [512][256]  ode_W0^T
constexpr size_t WS_W1T = WS_W0T + 512 * 256;     // [512][512]  ode_W1^T
constexpr size_t WS_W2T = WS_W1T + 512 * 512;     // [256][512]  ode_W2^T
constexpr size_t WS_WH0 = WS_W2T + 256 * 512;     // [768][256]  gru0_Wh^T
constexpr size_t WS_WX1 = WS_WH0 + 768 * 256;     // [768][256]  gru1_Wx^T
constexpr size_t WS_WH1 = WS_WX1 + 768 * 256;     // [768][256]  gru1_Wh^T

// ---------------- LDS pool layout (bytes) ----------------
constexpr int OFF_ADJ = 0;                        // f16 [80][96]
constexpr int OFF_H   = 15360;                    // f16 [256][80]  h (feature-major)
constexpr int OFF_H2  = 56320;                    // f16 [256][80]  h2 (feature-major)
constexpr int OFF_Z   = 97280;                    // f16 [512][80]  z (ODE) / rNM [80][256] (GRU)
constexpr int OFF_NM  = 179200;                   // f16 [80][512]  node-major scratch / ha+h2a
constexpr int OFF_OB0 = 261120;                   // f32[512]
constexpr int OFF_OB1 = 263168;                   // f32[512]
constexpr int OFF_OB2 = 265216;                   // f32[256]
constexpr int OFF_G0B = 266240;                   // f32[768]
constexpr int OFF_G1B = 269312;                   // f32[768]
constexpr int OFF_WX0 = 272384;                   // f32[2][768]
constexpr int OFF_OW  = 278528;                   // f32[256*3]
constexpr int OFF_OBO = 281600;                   // f32[4]
constexpr int OFF_XA  = 281616;                   // f32[80][2]
constexpr int OFF_XM  = 282256;                   // f32[80]
constexpr int OFF_XS  = 282576;                   // f32[80][2]
constexpr int POOLB   = 283280;                   // + tail pad

#define WMMA(a, b, c) \
  __builtin_amdgcn_wmma_f32_16x16x32_f16(false, (a), false, (b), (short)0, (c), false, false)

DEV v8f vzero() { v8f a = {0.f,0.f,0.f,0.f,0.f,0.f,0.f,0.f}; return a; }

// gfx1250 has V_TANH_F32 (TRANS32, co-executes with WMMA); use it when exposed.
DEV float fast_tanh(float x) {
#if __has_builtin(__builtin_amdgcn_tanhf)
  return __builtin_amdgcn_tanhf(x);
#else
  return tanhf(x);
#endif
}
DEV float sigmoidf_(float x) { return 0.5f * fast_tanh(0.5f * x) + 0.5f; }

// A-fragment: row m0+(l&15); halves {k0+sel*8..+7} U {k0+16+sel*8..+7}
DEV v16h load_a_frag(const _Float16* base, int lda, int m0, int k0, int lane) {
  int r = lane & 15, sel = lane >> 4;
  const _Float16* p = base + (size_t)(m0 + r) * lda + k0 + sel * 8;
  v8h lo = *(const v8h*)p;
  v8h hi = *(const v8h*)(p + 16);
  return __builtin_shufflevector(lo, hi, 0,1,2,3,4,5,6,7,8,9,10,11,12,13,14,15);
}

// B-fragment (matrix given as B^T row-major): row n0+(l&15); halves {k0+sel*8..+7} U {k0+16+sel*8..+7}
DEV v16h load_b_frag(const _Float16* base, int ldb, int n0, int k0, int lane) {
  int r = lane & 15, sel = lane >> 4;
  const _Float16* p = base + (size_t)(n0 + r) * ldb + k0 + sel * 8;
  v8h lo = *(const v8h*)p;
  v8h hi = *(const v8h*)(p + 16);
  return __builtin_shufflevector(lo, hi, 0,1,2,3,4,5,6,7,8,9,10,11,12,13,14,15);
}

// D tile transposed store: dst[n][m], one 16B store per lane
DEV void store_tr(_Float16* dst, int stride, int m0, int n0, v8f acc, int lane) {
  int r15 = lane & 15, sel = lane >> 4;
  _Float16* q = dst + (size_t)(n0 + r15) * stride + m0 + sel * 8;
  v8h o;
#pragma unroll
  for (int i = 0; i < 8; i++) o[i] = (_Float16)acc[i];
  *(v8h*)q = o;
}

// T^T[d][n] tiles = X^T · adj^T ; transposed-store node-major dst[n][d]
// adj B-fragments (15 of them) are loop-invariant: hoist into registers once.
DEV void gc_stage(const _Float16* __restrict__ Xt, int feat,
                  const _Float16* __restrict__ adjB, _Float16* __restrict__ dst,
                  int lane, int wave) {
  int MT = feat >> 4;
  v16h bf[15];
#pragma unroll
  for (int k = 0; k < 3; k++)
#pragma unroll
    for (int nt = 0; nt < 5; nt++)
      bf[k * 5 + nt] = load_b_frag(adjB, KADJ, nt * 16, k * 32, lane);
  for (int mt = wave; mt < MT; mt += 8) {
    int m0 = mt << 4;
    v8f acc[5] = {vzero(), vzero(), vzero(), vzero(), vzero()};
#pragma unroll
    for (int k = 0; k < 3; k++) {
      v16h a = load_a_frag(Xt, NP, m0, k * 32, lane);
#pragma unroll
      for (int nt = 0; nt < 5; nt++) acc[nt] = WMMA(a, bf[k * 5 + nt], acc[nt]);
    }
#pragma unroll
    for (int nt = 0; nt < 5; nt++) store_tr(dst, feat, m0, nt * 16, acc[nt], lane);
  }
}

// Z^T[o][n] = tanh(W^T · B_nm + b), scattered feature-major store (stride NP)
DEV void dense_tanh_stage(const _Float16* __restrict__ WT, int O, int K,
                          const _Float16* __restrict__ Bnm, const float* __restrict__ bias,
                          _Float16* __restrict__ dstT, int lane, int wave) {
  int MT = O >> 4, r15 = lane & 15, sel = lane >> 4;
  for (int mt = wave; mt < MT; mt += 8) {
    int m0 = mt << 4;
    v8f acc[5] = {vzero(), vzero(), vzero(), vzero(), vzero()};
    for (int k0 = 0; k0 < K; k0 += 32) {
      v16h a = load_a_frag(WT, K, m0, k0, lane);
#pragma unroll
      for (int nt = 0; nt < 5; nt++)
        acc[nt] = WMMA(a, load_b_frag(Bnm, K, nt * 16, k0, lane), acc[nt]);
    }
    int Mb = m0 + sel * 8;
#pragma unroll
    for (int nt = 0; nt < 5; nt++) {
      int n = nt * 16 + r15;
#pragma unroll
      for (int i = 0; i < 8; i++) {
        float v = fast_tanh(acc[nt][i] + bias[Mb + i]);
        dstT[(size_t)(Mb + i) * NP + n] = (_Float16)v;
      }
    }
  }
}

// h^T += DT * (W2^T · B_nm + b2)
DEV void dense_euler_stage(const _Float16* __restrict__ WT, int K,
                           const _Float16* __restrict__ Bnm, const float* __restrict__ bias,
                           _Float16* __restrict__ hT, int lane, int wave) {
  int r15 = lane & 15, sel = lane >> 4;
  for (int mt = wave; mt < 16; mt += 8) {
    int m0 = mt << 4;
    v8f acc[5] = {vzero(), vzero(), vzero(), vzero(), vzero()};
    for (int k0 = 0; k0 < K; k0 += 32) {
      v16h a = load_a_frag(WT, K, m0, k0, lane);
#pragma unroll
      for (int nt = 0; nt < 5; nt++)
        acc[nt] = WMMA(a, load_b_frag(Bnm, K, nt * 16, k0, lane), acc[nt]);
    }
    int Mb = m0 + sel * 8;
#pragma unroll
    for (int nt = 0; nt < 5; nt++) {
      int n = nt * 16 + r15;
#pragma unroll
      for (int i = 0; i < 8; i++) {
        size_t idx = (size_t)(Mb + i) * NP + n;
        float h = (float)hT[idx];
        hT[idx] = (_Float16)(h + DTC * (acc[nt][i] + bias[Mb + i]));
      }
    }
  }
}

// GRU0 r-gate: rNM[n][o] = sigmoid( (ha·Wh0)_r + gx_r ), transposed store
DEV void gru0_r_stage(const _Float16* __restrict__ Wh0T, const _Float16* __restrict__ ha,
                      const float* __restrict__ wx0, const float* __restrict__ g0b,
                      const float* __restrict__ xa, _Float16* __restrict__ rNM,
                      int lane, int wave) {
  int r15 = lane & 15, sel = lane >> 4;
  for (int mt = wave; mt < 16; mt += 8) {
    int m0 = mt << 4;
    v8f acc[5] = {vzero(), vzero(), vzero(), vzero(), vzero()};
    for (int k0 = 0; k0 < DR; k0 += 32) {
      v16h a = load_a_frag(Wh0T, DR, 256 + m0, k0, lane);
#pragma unroll
      for (int nt = 0; nt < 5; nt++)
        acc[nt] = WMMA(a, load_b_frag(ha, DR, nt * 16, k0, lane), acc[nt]);
    }
    int Mb = m0 + sel * 8;
#pragma unroll
    for (int nt = 0; nt < 5; nt++) {
      int n = nt * 16 + r15;
      float x0 = xa[n * 2], x1 = xa[n * 2 + 1];
      v8h o;
#pragma unroll
      for (int i = 0; i < 8; i++) {
        int oo = Mb + i;
        float g = g0b[256 + oo] + x0 * wx0[256 + oo] + x1 * wx0[768 + 256 + oo];
        o[i] = (_Float16)sigmoidf_(acc[nt][i] + g);
      }
      *(v8h*)(rNM + (size_t)n * DR + Mb) = o;
    }
  }
}

// GRU0 z/c gates + h2 = z*h1 + (1-z)*c -> h2^T (feature-major)
DEV void gru0_zc_stage(const _Float16* __restrict__ Wh0T, const _Float16* __restrict__ ha,
                       const float* __restrict__ wx0, const float* __restrict__ g0b,
                       const float* __restrict__ xa, const _Float16* __restrict__ rNM,
                       const _Float16* __restrict__ hT, _Float16* __restrict__ h2T,
                       int lane, int wave) {
  int r15 = lane & 15, sel = lane >> 4;
  for (int mt = wave; mt < 16; mt += 8) {
    int m0 = mt << 4;
    v8f aZ[5] = {vzero(), vzero(), vzero(), vzero(), vzero()};
    v8f aC[5] = {vzero(), vzero(), vzero(), vzero(), vzero()};
    for (int k0 = 0; k0 < DR; k0 += 32) {
      v16h az = load_a_frag(Wh0T, DR, m0, k0, lane);
      v16h ac = load_a_frag(Wh0T, DR, 512 + m0, k0, lane);
#pragma unroll
      for (int nt = 0; nt < 5; nt++) {
        v16h b = load_b_frag(ha, DR, nt * 16, k0, lane);
        aZ[nt] = WMMA(az, b, aZ[nt]);
        aC[nt] = WMMA(ac, b, aC[nt]);
      }
    }
    int Mb = m0 + sel * 8;
#pragma unroll
    for (int nt = 0; nt < 5; nt++) {
      int n = nt * 16 + r15;
      float x0 = xa[n * 2], x1 = xa[n * 2 + 1];
      v8h rv = *(const v8h*)(rNM + (size_t)n * DR + Mb);
#pragma unroll
      for (int i = 0; i < 8; i++) {
        int oo = Mb + i;
        float gz = g0b[oo] + x0 * wx0[oo] + x1 * wx0[768 + oo];
        float gc = g0b[512 + oo] + x0 * wx0[512 + oo] + x1 * wx0[768 + 512 + oo];
        float z = sigmoidf_(aZ[nt][i] + gz);
        float c = fast_tanh(gc + (float)rv[i] * aC[nt][i]);
        float h1v = (float)hT[(size_t)oo * NP + n];
        h2T[(size_t)oo * NP + n] = (_Float16)(z * h1v + (1.0f - z) * c);
      }
    }
  }
}

// GRU1 r-gate: rNM = sigmoid( h2a·Wx1_r + ha·Wh1_r + b_r )
DEV void gru1_r_stage(const _Float16* __restrict__ Wx1T, const _Float16* __restrict__ Wh1T,
                      const _Float16* __restrict__ h2a, const _Float16* __restrict__ ha,
                      const float* __restrict__ g1b, _Float16* __restrict__ rNM,
                      int lane, int wave) {
  int r15 = lane & 15, sel = lane >> 4;
  for (int mt = wave; mt < 16; mt += 8) {
    int m0 = mt << 4;
    v8f acc[5] = {vzero(), vzero(), vzero(), vzero(), vzero()};
    for (int k0 = 0; k0 < DR; k0 += 32) {
      v16h a1 = load_a_frag(Wx1T, DR, 256 + m0, k0, lane);
      v16h a2 = load_a_frag(Wh1T, DR, 256 + m0, k0, lane);
#pragma unroll
      for (int nt = 0; nt < 5; nt++) {
        acc[nt] = WMMA(a1, load_b_frag(h2a, DR, nt * 16, k0, lane), acc[nt]);
        acc[nt] = WMMA(a2, load_b_frag(ha,  DR, nt * 16, k0, lane), acc[nt]);
      }
    }
    int Mb = m0 + sel * 8;
#pragma unroll
    for (int nt = 0; nt < 5; nt++) {
      int n = nt * 16 + r15;
      v8h o;
#pragma unroll
      for (int i = 0; i < 8; i++)
        o[i] = (_Float16)sigmoidf_(acc[nt][i] + g1b[256 + Mb + i]);
      *(v8h*)(rNM + (size_t)n * DR + Mb) = o;
    }
  }
}

// GRU1 z/c + mask combine; writes h_new into hT in place
DEV void gru1_final_stage(const _Float16* __restrict__ Wx1T, const _Float16* __restrict__ Wh1T,
                          const _Float16* __restrict__ h2a, const _Float16* __restrict__ ha,
                          const float* __restrict__ g1b, const _Float16* __restrict__ rNM,
                          const float* __restrict__ xm, _Float16* __restrict__ hT,
                          int lane, int wave) {
  int r15 = lane & 15, sel = lane >> 4;
  for (int mt = wave; mt < 16; mt += 8) {
    int m0 = mt << 4;
    for (int nt = 0; nt < 5; nt++) {
      v8f accZ = vzero(), accC1 = vzero(), accC2 = vzero();
      for (int k0 = 0; k0 < DR; k0 += 32) {
        v16h b1 = load_b_frag(h2a, DR, nt * 16, k0, lane);
        v16h b2 = load_b_frag(ha,  DR, nt * 16, k0, lane);
        accZ  = WMMA(load_a_frag(Wx1T, DR, m0, k0, lane), b1, accZ);
        accZ  = WMMA(load_a_frag(Wh1T, DR, m0, k0, lane), b2, accZ);
        accC1 = WMMA(load_a_frag(Wx1T, DR, 512 + m0, k0, lane), b1, accC1);
        accC2 = WMMA(load_a_frag(Wh1T, DR, 512 + m0, k0, lane), b2, accC2);
      }
      int Mb = m0 + sel * 8;
      int n = nt * 16 + r15;
      float mval = xm[n];
      v8h rv = *(const v8h*)(rNM + (size_t)n * DR + Mb);
#pragma unroll
      for (int i = 0; i < 8; i++) {
        int oo = Mb + i;
        float z = sigmoidf_(accZ[i] + g1b[oo]);
        float c = fast_tanh(accC1[i] + g1b[512 + oo] + (float)rv[i] * accC2[i]);
        float h1v = (float)hT[(size_t)oo * NP + n];
        float h2p = z * h1v + (1.0f - z) * c;
        hT[(size_t)oo * NP + n] = (_Float16)(h1v * (1.0f - mval) + h2p * mval);
      }
    }
  }
}

// ---------------- weight prepass: f32 [K][O] -> f16 [O][K] ----------------
__global__ void transpose_f16_kernel(const float* __restrict__ src, _Float16* __restrict__ dst,
                                     int K, int O) {
  int idx = blockIdx.x * blockDim.x + threadIdx.x;
  if (idx < K * O) {
    int k = idx / O, o = idx % O;
    dst[(size_t)o * K + k] = (_Float16)src[idx];
  }
}

// ---------------- main recurrence: one block per batch element ----------------
__global__ __launch_bounds__(256) void gcodernn_kernel(
    const float* __restrict__ x2d, const float* __restrict__ maskp,
    const float* __restrict__ adj, const float* __restrict__ h0,
    const float* __restrict__ g0Wx, const float* __restrict__ g0bp,
    const float* __restrict__ g1bp, const float* __restrict__ ob0p,
    const float* __restrict__ ob1p, const float* __restrict__ ob2p,
    const float* __restrict__ outWp, const float* __restrict__ outbp,
    const _Float16* __restrict__ ws, float* __restrict__ out) {
  __shared__ __align__(16) unsigned char pool[POOLB];
  const int tid = threadIdx.x, lane = tid & 31, wave = tid >> 5;
  const int b = blockIdx.x;

  _Float16* adjB = (_Float16*)(pool + OFF_ADJ);
  _Float16* hT   = (_Float16*)(pool + OFF_H);
  _Float16* h2T  = (_Float16*)(pool + OFF_H2);
  _Float16* zT   = (_Float16*)(pool + OFF_Z);
  _Float16* rNM  = (_Float16*)(pool + OFF_Z);          // GRU reuse of zT space
  _Float16* bnm  = (_Float16*)(pool + OFF_NM);
  _Float16* ha   = (_Float16*)(pool + OFF_NM);          // [80][256]
  _Float16* h2a  = (_Float16*)(pool + OFF_NM) + 80 * DR;
  float* ob0 = (float*)(pool + OFF_OB0);
  float* ob1 = (float*)(pool + OFF_OB1);
  float* ob2 = (float*)(pool + OFF_OB2);
  float* g0b = (float*)(pool + OFF_G0B);
  float* g1b = (float*)(pool + OFF_G1B);
  float* wx0 = (float*)(pool + OFF_WX0);
  float* oW  = (float*)(pool + OFF_OW);
  float* oB  = (float*)(pool + OFF_OBO);
  float* xa  = (float*)(pool + OFF_XA);
  float* xm  = (float*)(pool + OFF_XM);
  float* xsb = (float*)(pool + OFF_XS);

  const _Float16* W0T  = ws + WS_W0T;
  const _Float16* W1T  = ws + WS_W1T;
  const _Float16* W2T  = ws + WS_W2T;
  const _Float16* Wh0T = ws + WS_WH0;
  const _Float16* Wx1T = ws + WS_WX1;
  const _Float16* Wh1T = ws + WS_WH1;

  // zero entire pool (keeps padded regions & all WMMA over-reads finite)
  for (int i = tid; i < POOLB / 4; i += 256) ((int*)pool)[i] = 0;
  __syncthreads();

  for (int i = tid; i < NN * NN; i += 256) {
    int n = i / NN, m = i % NN;
    adjB[n * KADJ + m] = (_Float16)adj[(size_t)b * NN * NN + i];
  }
  for (int i = tid; i < DH; i += 256) { ob0[i] = ob0p[i]; ob1[i] = ob1p[i]; }
  for (int i = tid; i < DR; i += 256) ob2[i] = ob2p[i];
  for (int i = tid; i < G3; i += 256) { g0b[i] = g0bp[i]; g1b[i] = g1bp[i]; }
  for (int i = tid; i < 2 * G3; i += 256) wx0[i] = g0Wx[i];
  for (int i = tid; i < DR * 3; i += 256) oW[i] = outWp[i];
  if (tid < 3) oB[tid] = outbp[tid];
  for (int i = tid; i < DR * NP; i += 256) {
    int d = i / NP, n = i % NP;
    hT[i] = (n < NN) ? (_Float16)h0[d] : (_Float16)0.0f;
  }
  __syncthreads();

  for (int s = 0; s < SS; s++) {
    // prefetch next step's tiny global inputs behind this step's compute
    if (s + 1 < SS && tid < NN) {
      __builtin_prefetch(&x2d[(((size_t)b * SS + s + 1) * NN + tid) * 2], 0, 0);
      __builtin_prefetch(&maskp[((size_t)b * SS + s + 1) * NN + tid], 0, 0);
    }

    // ---- masked input + xa = adj @ xs (tiny K=2, VALU) ----
    if (tid < NP) {
      float m = (tid < NN) ? maskp[((size_t)b * SS + s) * NN + tid] : 0.0f;
      xm[tid] = m;
      float v0 = 0.f, v1 = 0.f;
      if (tid < NN) {
        size_t base = (((size_t)b * SS + s) * NN + tid) * 2;
        v0 = x2d[base] * m;
        v1 = x2d[base + 1] * m;
      }
      xsb[tid * 2] = v0;
      xsb[tid * 2 + 1] = v1;
    }
    __syncthreads();
    if (tid < NP) {
      float a0 = 0.f, a1 = 0.f;
      if (tid < NN)
        for (int m = 0; m < NN; m++) {
          float w = (float)adjB[tid * KADJ + m];
          a0 += w * xsb[m * 2];
          a1 += w * xsb[m * 2 + 1];
        }
      xa[tid * 2] = a0;
      xa[tid * 2 + 1] = a1;
    }

    // ---- ODE: 4 explicit Euler steps of the 3-layer GNN ----
    for (int it = 0; it < 4; it++) {
      gc_stage(hT, DR, adjB, bnm, lane, wave);                       // [80][256]
      __syncthreads();
      dense_tanh_stage(W0T, DH, DR, bnm, ob0, zT, lane, wave);       // z1^T [512][80]
      __syncthreads();
      gc_stage(zT, DH, adjB, bnm, lane, wave);                       // [80][512]
      __syncthreads();
      dense_tanh_stage(W1T, DH, DH, bnm, ob1, zT, lane, wave);       // z2^T
      __syncthreads();
      gc_stage(zT, DH, adjB, bnm, lane, wave);                       // [80][512]
      __syncthreads();
      dense_euler_stage(W2T, DH, bnm, ob2, hT, lane, wave);          // h += DT*dh
      __syncthreads();
    }

    // ---- per-step output ys = h1 @ out_W + out_b ----
    if (tid < NN * 3) {
      int n = tid / 3, o = tid % 3;
      float acc = oB[o];
      for (int d = 0; d < DR; d++) acc += (float)hT[(size_t)d * NP + n] * oW[d * 3 + o];
      out[(((size_t)b * SS + s) * NN + n) * 3 + o] = acc;
    }

    // ---- GRU0: h2 = gcgru(x, h1) ----
    gc_stage(hT, DR, adjB, ha, lane, wave);                          // ha = gc(adj,h1)
    __syncthreads();
    gru0_r_stage(Wh0T, ha, wx0, g0b, xa, rNM, lane, wave);
    __syncthreads();
    gru0_zc_stage(Wh0T, ha, wx0, g0b, xa, rNM, hT, h2T, lane, wave);
    __syncthreads();

    // ---- GRU1: h2' = gcgru(h2, h1); h = h1*(1-m) + h2'*m ----
    gc_stage(h2T, DR, adjB, h2a, lane, wave);                        // gc(adj,h2)
    __syncthreads();
    gru1_r_stage(Wx1T, Wh1T, h2a, ha, g1b, rNM, lane, wave);
    __syncthreads();
    gru1_final_stage(Wx1T, Wh1T, h2a, ha, g1b, rNM, xm, hT, lane, wave);
    __syncthreads();
  }
}

extern "C" void kernel_launch(void* const* d_in, const int* in_sizes, int n_in,
                              void* d_out, int out_size, void* d_ws, size_t ws_size,
                              hipStream_t stream) {
  const float* x2d   = (const float*)d_in[0];
  const float* mask  = (const float*)d_in[1];
  const float* adj   = (const float*)d_in[2];
  const float* h0    = (const float*)d_in[3];
  const float* g0Wx  = (const float*)d_in[4];
  const float* g0Wh  = (const float*)d_in[5];
  const float* g0b   = (const float*)d_in[6];
  const float* g1Wx  = (const float*)d_in[7];
  const float* g1Wh  = (const float*)d_in[8];
  const float* g1b   = (const float*)d_in[9];
  const float* oW0   = (const float*)d_in[10];
  const float* ob0   = (const float*)d_in[11];
  const float* oW1   = (const float*)d_in[12];
  const float* ob1   = (const float*)d_in[13];
  const float* oW2   = (const float*)d_in[14];
  const float* ob2   = (const float*)d_in[15];
  const float* outW  = (const float*)d_in[16];
  const float* outb  = (const float*)d_in[17];
  _Float16* ws = (_Float16*)d_ws;
  float* out = (float*)d_out;

  auto tgrid = [](int e) { return dim3((e + 255) / 256); };
  transpose_f16_kernel<<<tgrid(256 * 512), 256, 0, stream>>>(oW0,  ws + WS_W0T, 256, 512);
  transpose_f16_kernel<<<tgrid(512 * 512), 256, 0, stream>>>(oW1,  ws + WS_W1T, 512, 512);
  transpose_f16_kernel<<<tgrid(512 * 256), 256, 0, stream>>>(oW2,  ws + WS_W2T, 512, 256);
  transpose_f16_kernel<<<tgrid(256 * 768), 256, 0, stream>>>(g0Wh, ws + WS_WH0, 256, 768);
  transpose_f16_kernel<<<tgrid(256 * 768), 256, 0, stream>>>(g1Wx, ws + WS_WX1, 256, 768);
  transpose_f16_kernel<<<tgrid(256 * 768), 256, 0, stream>>>(g1Wh, ws + WS_WH1, 256, 768);

  gcodernn_kernel<<<dim3(BB), dim3(256), 0, stream>>>(
      x2d, mask, adj, h0, g0Wx, g0b, g1b, ob0, ob1, ob2, outW, outb, ws, out);
}
